// GINet_10428180595432
// MI455X (gfx1250) — matrix operand order repaired and checked by reading, not compiled, not run on hardware
//
#include <hip/hip_runtime.h>
#include <hip/hip_bf16.h>
#include <math.h>

// ---------------------------------------------------------------------------
// Types
// ---------------------------------------------------------------------------
typedef __attribute__((ext_vector_type(16))) __bf16 bf16x16;
typedef __attribute__((ext_vector_type(8)))  __bf16 bf16x8;
typedef __attribute__((ext_vector_type(4)))  __bf16 bf16x4;
typedef __attribute__((ext_vector_type(8)))  float  f32x8;

#define EMB        300
#define EMBP       320   // EMB padded to mult of 32
#define HID        600
#define HIDP       608   // HID padded to mult of 32
#define FEATD      512
#define HFEAT      256
#define BN_EPS     1e-5f

#define ACT_NONE      0
#define ACT_RELU      1
#define ACT_SOFTPLUS  2

__device__ __forceinline__ void atomicAddF(float* p, float v) {
  __hip_atomic_fetch_add(p, v, __ATOMIC_RELAXED, __HIP_MEMORY_SCOPE_AGENT);
}

// ---------------------------------------------------------------------------
// bf16 WMMA GEMM:  C[M,N] = act(A[M,Kp] @ Bt^T + bias)
//  - A: f32 or bf16, leading dim lda (Kp = ktiles*32 <= lda), pad-K cols are 0
//  - Bt: packed bf16, TRANSPOSED [Np][Kp], pad entries 0  (ldb = Kp)
//  - C: f32 or bf16, leading dim ldc; rows padded (no M guard), col guard vs N
// Block 256 threads = 8 waves; tile 128x64, K-step 32; wave = 2x2 WMMA tiles.
// Software pipelined: fetch tile kt+1 to regs while computing tile kt.
// ---------------------------------------------------------------------------
#define BM   128
#define BNC  64
#define BKK  32
#define LSTR 40   // LDS row stride in bf16 (80B -> every row 16B aligned)

template<int ACT, typename AT, typename CT>
__global__ __launch_bounds__(256) void gemm_bias_act(
    const AT* __restrict__ A, int lda,
    const __bf16* __restrict__ Bt, int ldb,
    const float* __restrict__ bias,
    CT* __restrict__ C, int ldc,
    int ktiles, int N)
{
  __shared__ alignas(16) __bf16 As[BM  * LSTR];
  __shared__ alignas(16) __bf16 Bs[BNC * LSTR];

  const int tid  = threadIdx.x;
  const int lane = tid & 31;
  const int wave = tid >> 5;
  const int wm   = wave >> 1;       // 0..3
  const int wn   = wave & 1;        // 0..1
  const int row0 = blockIdx.y * BM;
  const int col0 = blockIdx.x * BNC;
  const int lr   = lane & 15;
  const int hi   = lane >> 4;       // 0 or 1

  // staging maps
  const int arF = tid >> 3;            // f32 A: 32 rows x (8 thr * float4)
  const int acF = (tid & 7) * 4;
  const int arH = tid >> 1;            // bf16 A: 128 rows x (2 thr * 16 bf16)
  const int acH = (tid & 1) * 16;
  const int bn  = tid & 63;            // B: 64 cols x (4 thr * 8 bf16)
  const int bko = (tid >> 6) * 8;

  float4 aRegF[4];
  bf16x8 aRegH[2];
  bf16x8 bReg;

  auto fetch = [&](int kt) {
    const int k0 = kt * BKK;
    if constexpr (sizeof(AT) == 4) {
      const float* ap = (const float*)A + (size_t)(row0 + arF) * lda + k0 + acF;
      #pragma unroll
      for (int rr = 0; rr < 4; ++rr)
        aRegF[rr] = *(const float4*)(ap + (size_t)rr * 32 * lda);
    } else {
      const __bf16* ap = (const __bf16*)A + (size_t)(row0 + arH) * lda + k0 + acH;
      aRegH[0] = *(const bf16x8*)(ap);
      aRegH[1] = *(const bf16x8*)(ap + 8);
    }
    bReg = *(const bf16x8*)(Bt + (size_t)(col0 + bn) * ldb + k0 + bko);
  };
  auto stage = [&]() {
    if constexpr (sizeof(AT) == 4) {
      #pragma unroll
      for (int rr = 0; rr < 4; ++rr) {
        bf16x4 w;
        w.x = (__bf16)aRegF[rr].x; w.y = (__bf16)aRegF[rr].y;
        w.z = (__bf16)aRegF[rr].z; w.w = (__bf16)aRegF[rr].w;
        *(bf16x4*)&As[(arF + rr * 32) * LSTR + acF] = w;
      }
    } else {
      *(bf16x8*)&As[arH * LSTR + acH]     = aRegH[0];
      *(bf16x8*)&As[arH * LSTR + acH + 8] = aRegH[1];
    }
    *(bf16x8*)&Bs[bn * LSTR + bko] = bReg;
  };

  f32x8 acc[2][2] = {};
  fetch(0);

  for (int kt = 0; kt < ktiles; ++kt) {
    stage();
    __syncthreads();
    if (kt + 1 < ktiles) fetch(kt + 1);

    // A 16x32 frag: lane(0-15)=row, K chunks at kbase and kbase+16, kbase=hi*8
    bf16x16 afr[2];
    const int akb = hi * 8;
    #pragma unroll
    for (int tm = 0; tm < 2; ++tm) {
      const int r = wm * 32 + tm * 16 + lr;
      union { bf16x16 v; bf16x8 h[2]; } u;
      u.h[0] = *(const bf16x8*)&As[r * LSTR + akb];
      u.h[1] = *(const bf16x8*)&As[r * LSTR + akb + 16];
      afr[tm] = u.v;
    }
    // B 32x16 frag: lane(0-15)=col, K chunks at kbase and kbase+16, kbase=hi*8
    bf16x16 bfr[2];
    const int bkb = hi * 8;
    #pragma unroll
    for (int tn = 0; tn < 2; ++tn) {
      const int c = wn * 32 + tn * 16 + lr;
      union { bf16x16 v; bf16x8 h[2]; } u;
      u.h[0] = *(const bf16x8*)&Bs[c * LSTR + bkb];
      u.h[1] = *(const bf16x8*)&Bs[c * LSTR + bkb + 16];
      bfr[tn] = u.v;
    }

    #pragma unroll
    for (int tm = 0; tm < 2; ++tm)
      #pragma unroll
      for (int tn = 0; tn < 2; ++tn)
        acc[tm][tn] = __builtin_amdgcn_wmma_f32_16x16x32_bf16(
            false, afr[tm], false, bfr[tn], (short)0, acc[tm][tn], false, false);

    __syncthreads();
  }

  // epilogue: bias + activation (rows padded -> no M guard; col guard vs N)
  #pragma unroll
  for (int tn = 0; tn < 2; ++tn) {
    const int col = col0 + wn * 32 + tn * 16 + lr;
    if (col >= N) continue;
    const float bv = bias[col];
    #pragma unroll
    for (int tm = 0; tm < 2; ++tm) {
      const size_t rbase = (size_t)(row0 + wm * 32 + tm * 16 + hi * 8);
      #pragma unroll
      for (int r = 0; r < 8; ++r) {
        float o = acc[tm][tn][r] + bv;
        if (ACT == ACT_RELU)          o = fmaxf(o, 0.0f);
        else if (ACT == ACT_SOFTPLUS) o = (o > 20.0f) ? o : log1pf(expf(o));
        C[(rbase + r) * (size_t)ldc + col] = (CT)o;
      }
    }
  }
}

// ---------------------------------------------------------------------------
// Weight packing: Bt[n*KP + k] = bf16(B[k*N + n]), zero-padded to NP x KP
// ---------------------------------------------------------------------------
__global__ void pack_bt_kernel(const float* __restrict__ B, __bf16* __restrict__ Bt,
                               int K, int N, int KP, int NP)
{
  size_t idx = (size_t)blockIdx.x * blockDim.x + threadIdx.x;
  size_t total = (size_t)NP * KP;
  if (idx >= total) return;
  int n = (int)(idx / KP);
  int k = (int)(idx - (size_t)n * KP);
  float v = (n < N && k < K) ? B[(size_t)k * N + n] : 0.0f;
  Bt[idx] = (__bf16)v;
}

__global__ void zero_padcols_bf16_kernel(__bf16* __restrict__ buf, int ld,
                                         int c0, int rows)
{
  int w = ld - c0;
  size_t idx = (size_t)blockIdx.x * blockDim.x + threadIdx.x;
  size_t total = (size_t)rows * w;
  if (idx >= total) return;
  int r = (int)(idx / w);
  int c = c0 + (int)(idx - (size_t)r * w);
  buf[(size_t)r * ld + c] = (__bf16)0.0f;
}

// ---------------------------------------------------------------------------
// Graph / elementwise kernels (float4 vectorized; EMB=300 % 4 == 0)
// ---------------------------------------------------------------------------
__global__ void embed_kernel(const int* __restrict__ x,
                             const float* __restrict__ emb1,
                             const float* __restrict__ emb2,
                             float* __restrict__ h, int n_nodes)
{
  size_t idx = (size_t)blockIdx.x * blockDim.x + threadIdx.x;   // Nn * 75
  size_t total = (size_t)n_nodes * (EMB / 4);
  if (idx >= total) return;
  int n = (int)(idx / (EMB / 4));
  int c = (int)(idx - (size_t)n * (EMB / 4)) * 4;
  const float4 a = *(const float4*)(emb1 + (size_t)x[n*2+0] * EMB + c);
  const float4 b = *(const float4*)(emb2 + (size_t)x[n*2+1] * EMB + c);
  float4 o; o.x = a.x + b.x; o.y = a.y + b.y; o.z = a.z + b.z; o.w = a.w + b.w;
  *(float4*)(h + (size_t)n * EMB + c) = o;
}

// agg[n, 0..319] = (c<300) ? h[n,c] + e_self[c] : 0   (self loop attr = (4,0))
__global__ void init_agg_kernel(const float* __restrict__ h,
                                const float* __restrict__ e1,
                                const float* __restrict__ e2,
                                float* __restrict__ agg, int n_nodes)
{
  size_t idx = (size_t)blockIdx.x * blockDim.x + threadIdx.x;   // Nn * 80
  size_t total = (size_t)n_nodes * (EMBP / 4);
  if (idx >= total) return;
  int n = (int)(idx / (EMBP / 4));
  int c = (int)(idx - (size_t)n * (EMBP / 4)) * 4;
  float4 o = {0.0f, 0.0f, 0.0f, 0.0f};
  if (c < EMB) {
    const float4 hv = *(const float4*)(h + (size_t)n * EMB + c);
    const float4 a  = *(const float4*)(e1 + 4 * EMB + c);
    const float4 b  = *(const float4*)(e2 + c);
    o.x = hv.x + a.x + b.x; o.y = hv.y + a.y + b.y;
    o.z = hv.z + a.z + b.z; o.w = hv.w + a.w + b.w;
  }
  *(float4*)(agg + (size_t)n * EMBP + c) = o;
}

__global__ void scatter_kernel(const float* __restrict__ h,
                               const int* __restrict__ src,
                               const int* __restrict__ dst,
                               const int* __restrict__ ea,
                               const float* __restrict__ e1,
                               const float* __restrict__ e2,
                               float* __restrict__ agg, int n_edges)
{
  size_t idx = (size_t)blockIdx.x * blockDim.x + threadIdx.x;   // Ne * 75
  size_t total = (size_t)n_edges * (EMB / 4);
  if (idx >= total) return;
  int e = (int)(idx / (EMB / 4));
  int c = (int)(idx - (size_t)e * (EMB / 4)) * 4;
  int s = src[e], d = dst[e];
  int a0 = ea[e*2+0], a1 = ea[e*2+1];
  const float4 hv = *(const float4*)(h  + (size_t)s  * EMB + c);
  const float4  av = *(const float4*)(e1 + (size_t)a0 * EMB + c);
  const float4 bv = *(const float4*)(e2 + (size_t)a1 * EMB + c);
  float* out = agg + (size_t)d * EMBP + c;
  atomicAddF(out + 0, hv.x + av.x + bv.x);
  atomicAddF(out + 1, hv.y + av.y + bv.y);
  atomicAddF(out + 2, hv.z + av.z + bv.z);
  atomicAddF(out + 3, hv.w + av.w + bv.w);
}

// Deterministic two-stage BatchNorm reduction over h [Nn, EMB] (ld=EMB)
__global__ void bn_partial_kernel(const float* __restrict__ h,
                                  float* __restrict__ partials,
                                  int n_nodes, int rows_per_chunk)
{
  int c = threadIdx.x;
  if (c >= EMB) return;
  int r0 = blockIdx.x * rows_per_chunk;
  int r1 = r0 + rows_per_chunk;
  if (r1 > n_nodes) r1 = n_nodes;
  float s = 0.0f, q = 0.0f;
  for (int r = r0; r < r1; ++r) {
    float v = h[(size_t)r * EMB + c];
    s += v; q += v * v;
  }
  partials[(size_t)blockIdx.x * 2 * EMB + c]       = s;
  partials[(size_t)blockIdx.x * 2 * EMB + EMB + c] = q;
}

__global__ void bn_finalize_kernel(const float* __restrict__ partials,
                                   const float* __restrict__ gamma,
                                   const float* __restrict__ beta,
                                   float* __restrict__ scale,
                                   float* __restrict__ shift,
                                   int nchunks, int n_nodes)
{
  int c = threadIdx.x;
  if (c >= EMB) return;
  float s = 0.0f, q = 0.0f;
  for (int b = 0; b < nchunks; ++b) {
    s += partials[(size_t)b * 2 * EMB + c];
    q += partials[(size_t)b * 2 * EMB + EMB + c];
  }
  float inv_n = 1.0f / (float)n_nodes;
  float mu  = s * inv_n;
  float var = q * inv_n - mu * mu;
  float sc  = gamma[c] * rsqrtf(var + BN_EPS);
  scale[c] = sc;
  shift[c] = beta[c] - mu * sc;
}

__global__ void bn_apply_kernel(float* __restrict__ h,
                                const float* __restrict__ scale,
                                const float* __restrict__ shift,
                                int n_nodes, int relu)
{
  size_t idx = (size_t)blockIdx.x * blockDim.x + threadIdx.x;   // Nn * 75
  size_t total = (size_t)n_nodes * (EMB / 4);
  if (idx >= total) return;
  int n = (int)(idx / (EMB / 4));
  int c = (int)(idx - (size_t)n * (EMB / 4)) * 4;
  float4 v = *(float4*)(h + (size_t)n * EMB + c);
  const float4 sc = *(const float4*)(scale + c);
  const float4 sh = *(const float4*)(shift + c);
  v.x = v.x * sc.x + sh.x; v.y = v.y * sc.y + sh.y;
  v.z = v.z * sc.z + sh.z; v.w = v.w * sc.w + sh.w;
  if (relu) {
    v.x = fmaxf(v.x, 0.0f); v.y = fmaxf(v.y, 0.0f);
    v.z = fmaxf(v.z, 0.0f); v.w = fmaxf(v.w, 0.0f);
  }
  *(float4*)(h + (size_t)n * EMB + c) = v;
}

__global__ void pool_sum_kernel(const float* __restrict__ h,
                                const int* __restrict__ batch,
                                float* __restrict__ hg, int n_nodes)
{
  size_t idx = (size_t)blockIdx.x * blockDim.x + threadIdx.x;   // Nn * 75
  size_t total = (size_t)n_nodes * (EMB / 4);
  if (idx >= total) return;
  int n = (int)(idx / (EMB / 4));
  int c = (int)(idx - (size_t)n * (EMB / 4)) * 4;
  const float4 v = *(const float4*)(h + (size_t)n * EMB + c);
  float* out = hg + (size_t)batch[n] * EMBP + c;
  atomicAddF(out + 0, v.x); atomicAddF(out + 1, v.y);
  atomicAddF(out + 2, v.z); atomicAddF(out + 3, v.w);
}

__global__ void pool_cnt_kernel(const int* __restrict__ batch,
                                float* __restrict__ cnt, int n_nodes)
{
  int n = blockIdx.x * blockDim.x + threadIdx.x;
  if (n >= n_nodes) return;
  atomicAddF(&cnt[batch[n]], 1.0f);
}

__global__ void pool_div_kernel(float* __restrict__ hg,
                                const float* __restrict__ cnt, int g)
{
  size_t idx = (size_t)blockIdx.x * blockDim.x + threadIdx.x;   // G * 75
  size_t total = (size_t)g * (EMB / 4);
  if (idx >= total) return;
  int gi = (int)(idx / (EMB / 4));
  int c = (int)(idx - (size_t)gi * (EMB / 4)) * 4;
  float inv = 1.0f / fmaxf(cnt[gi], 1.0f);
  float4 v = *(float4*)(hg + (size_t)gi * EMBP + c);
  v.x *= inv; v.y *= inv; v.z *= inv; v.w *= inv;
  *(float4*)(hg + (size_t)gi * EMBP + c) = v;
}

__global__ void head_kernel(const float* __restrict__ z,
                            const float* __restrict__ w,
                            const float* __restrict__ b,
                            float* __restrict__ out, int g)
{
  int gi = blockIdx.x * blockDim.x + threadIdx.x;
  if (gi >= g) return;
  float s = b[0];
  const float* zr = z + (size_t)gi * HFEAT;
  #pragma unroll 4
  for (int c = 0; c < HFEAT; ++c) s += zr[c] * w[c];
  out[gi] = s;
}

// ---------------------------------------------------------------------------
// Launch
// ---------------------------------------------------------------------------
extern "C" void kernel_launch(void* const* d_in, const int* in_sizes, int n_in,
                              void* d_out, int out_size, void* d_ws, size_t ws_size,
                              hipStream_t stream)
{
  // setup_inputs() dict order (recursive insertion order)
  const int*   x          = (const int*)d_in[0];
  const int*   edge_index = (const int*)d_in[1];
  const int*   edge_attr  = (const int*)d_in[2];
  const int*   batch      = (const int*)d_in[3];
  const float* x_emb1     = (const float*)d_in[4];
  const float* x_emb2     = (const float*)d_in[5];
  // layers: 6 + 8*l : {e1,e2,W1,b1,W2,b2,gamma,beta}
  const float* feat_W = (const float*)d_in[46];
  const float* feat_b = (const float*)d_in[47];
  const float* p0_W   = (const float*)d_in[48];
  const float* p0_b   = (const float*)d_in[49];
  const float* p1_W   = (const float*)d_in[50];
  const float* p1_b   = (const float*)d_in[51];
  const float* p2_W   = (const float*)d_in[52];
  const float* p2_b   = (const float*)d_in[53];

  const int Nn = in_sizes[3];
  const int Ne = in_sizes[2] / 2;
  const int Gg = out_size;
  const int Mpad = ((Nn + BM - 1) / BM) * BM;   // padded node-row count

  // workspace carve-out (float units)
  float* ws = (float*)d_ws;
  size_t off = 0;
  auto carve = [&](size_t nelem) { float* p = ws + off; off += (nelem + 63) & ~(size_t)63; return p; };
  float*  h      = carve((size_t)Mpad * EMB);
  float*  agg    = carve((size_t)Mpad * EMBP);
  __bf16* hidden = (__bf16*)carve(((size_t)Mpad * HIDP + 1) / 2);
  const int RPC = 512;
  const int nchunks = (Nn + RPC - 1) / RPC;
  float* partials = carve((size_t)nchunks * 2 * EMB);
  float* bn_scale = carve(EMB);
  float* bn_shift = carve(EMB);
  float* hg   = carve((size_t)Gg * EMBP);
  float* cnt  = carve(Gg);
  float* fbuf = carve((size_t)Gg * FEATD);
  float* z0   = carve((size_t)Gg * HFEAT);
  float* z1   = carve((size_t)Gg * HFEAT);
  __bf16* btS = (__bf16*)carve(102400);          // packed-weight scratch (<=640*320 bf16)
  (void)ws_size; (void)n_in;

  const int* src = edge_index;
  const int* dst = edge_index + Ne;

  const int gN4  = (int)(((size_t)Nn * (EMB / 4)  + 255) / 256);
  const int gN4p = (int)(((size_t)Nn * (EMBP / 4) + 255) / 256);
  const int gE4  = (int)(((size_t)Ne * (EMB / 4)  + 255) / 256);

  embed_kernel<<<gN4, 256, 0, stream>>>(x, x_emb1, x_emb2, h, Nn);

  const dim3 gGemm1((HID + BNC - 1) / BNC, Mpad / BM);   // 10 x Mpad/128
  const dim3 gGemm2((EMB + BNC - 1) / BNC, Mpad / BM);   // 5  x Mpad/128

  for (int l = 0; l < 5; ++l) {
    const float* e1    = (const float*)d_in[6 + 8*l + 0];
    const float* e2    = (const float*)d_in[6 + 8*l + 1];
    const float* W1    = (const float*)d_in[6 + 8*l + 2];
    const float* b1    = (const float*)d_in[6 + 8*l + 3];
    const float* W2    = (const float*)d_in[6 + 8*l + 4];
    const float* b2    = (const float*)d_in[6 + 8*l + 5];
    const float* gamma = (const float*)d_in[6 + 8*l + 6];
    const float* beta  = (const float*)d_in[6 + 8*l + 7];

    // message passing: agg = h + e_self (padded cols zeroed), += edge messages
    init_agg_kernel<<<gN4p, 256, 0, stream>>>(h, e1, e2, agg, Nn);
    scatter_kernel<<<gE4, 256, 0, stream>>>(h, src, dst, edge_attr, e1, e2, agg, Ne);

    // hidden(bf16) = relu(agg @ W1 + b1)
    pack_bt_kernel<<<(640*320 + 255) / 256, 256, 0, stream>>>(W1, btS, EMB, HID, EMBP, 640);
    gemm_bias_act<ACT_RELU, float, __bf16><<<gGemm1, 256, 0, stream>>>(
        agg, EMBP, btS, EMBP, b1, hidden, HIDP, EMBP / BKK, HID);
    zero_padcols_bf16_kernel<<<((size_t)Mpad * (HIDP - HID) + 255) / 256, 256, 0, stream>>>(
        hidden, HIDP, HID, Mpad);

    // h = hidden @ W2 + b2
    pack_bt_kernel<<<(320*608 + 255) / 256, 256, 0, stream>>>(W2, btS, HID, EMB, HIDP, EMBP);
    gemm_bias_act<ACT_NONE, __bf16, float><<<gGemm2, 256, 0, stream>>>(
        hidden, HIDP, btS, HIDP, b2, h, EMB, HIDP / BKK, EMB);

    // BatchNorm (+ReLU except last layer)
    bn_partial_kernel<<<nchunks, 320, 0, stream>>>(h, partials, Nn, RPC);
    bn_finalize_kernel<<<1, 320, 0, stream>>>(partials, gamma, beta, bn_scale, bn_shift, nchunks, Nn);
    bn_apply_kernel<<<gN4, 256, 0, stream>>>(h, bn_scale, bn_shift, Nn, (l < 4) ? 1 : 0);
  }

  // mean pooling per graph (hg padded to EMBP, pad cols stay 0 from memset)
  hipMemsetAsync(hg, 0, (size_t)Gg * EMBP * sizeof(float), stream);
  hipMemsetAsync(cnt, 0, (size_t)Gg * sizeof(float), stream);
  pool_sum_kernel<<<gN4, 256, 0, stream>>>(h, batch, hg, Nn);
  pool_cnt_kernel<<<(Nn + 255) / 256, 256, 0, stream>>>(batch, cnt, Nn);
  pool_div_kernel<<<(int)(((size_t)Gg * (EMB / 4) + 255) / 256), 256, 0, stream>>>(hg, cnt, Gg);

  // prediction head
  pack_bt_kernel<<<(512*320 + 255) / 256, 256, 0, stream>>>(feat_W, btS, EMB, FEATD, EMBP, FEATD);
  dim3 gf(FEATD / BNC, Gg / BM);
  gemm_bias_act<ACT_NONE, float, float><<<gf, 256, 0, stream>>>(
      hg, EMBP, btS, EMBP, feat_b, fbuf, FEATD, EMBP / BKK, FEATD);

  pack_bt_kernel<<<(256*512 + 255) / 256, 256, 0, stream>>>(p0_W, btS, FEATD, HFEAT, FEATD, HFEAT);
  dim3 gp(HFEAT / BNC, Gg / BM);
  gemm_bias_act<ACT_SOFTPLUS, float, float><<<gp, 256, 0, stream>>>(
      fbuf, FEATD, btS, FEATD, p0_b, z0, HFEAT, FEATD / BKK, HFEAT);

  pack_bt_kernel<<<(256*256 + 255) / 256, 256, 0, stream>>>(p1_W, btS, HFEAT, HFEAT, HFEAT, HFEAT);
  gemm_bias_act<ACT_SOFTPLUS, float, float><<<gp, 256, 0, stream>>>(
      z0, HFEAT, btS, HFEAT, p1_b, z1, HFEAT, HFEAT / BKK, HFEAT);

  head_kernel<<<(Gg + 255) / 256, 256, 0, stream>>>(z1, p2_W, p2_b, (float*)d_out, Gg);
}
